// FrontAttention_55198919688795
// MI455X (gfx1250) — matrix operand order repaired and checked by reading, not compiled, not run on hardware
//
#include <hip/hip_runtime.h>

// ---------------- types ----------------
typedef __bf16 bf16;
typedef __attribute__((ext_vector_type(16))) __bf16 bf16x16;
typedef __attribute__((ext_vector_type(8)))  __bf16 bf16x8;
typedef __attribute__((ext_vector_type(8)))  float  f32x8;

// ---------------- problem constants ----------------
constexpr int C_  = 384;
constexpr int N_  = 4096;
constexpr int NH_ = 8;
constexpr int HD_ = 48;     // head dim
constexpr int CH_ = 64;     // chunk (local attention window)
constexpr int NC_ = 64;     // N / CH
constexpr int NTHREADS = 288;     // 9 waves of 32
constexpr int LDX = C_ + 8;       // padded K-contiguous stride (16B-aligned groups)

// ---------------- WMMA fragment helpers ----------------
__device__ __forceinline__ bf16x16 cat8(bf16x8 lo, bf16x8 hi) {
  return __builtin_shufflevector(lo, hi, 0,1,2,3,4,5,6,7,8,9,10,11,12,13,14,15);
}
// A fragment (16x32 bf16): row-major [M][K]; caller passes &A[row][kb].
// elements 0..7  <- K = kb + half*8 + e
// elements 8..15 <- K = kb + 16 + half*8 + (e-8)
__device__ __forceinline__ bf16x16 fragA(const bf16* rowk, int half) {
  bf16x8 lo = *(const bf16x8*)(rowk + half * 8);
  bf16x8 hi = *(const bf16x8*)(rowk + 16 + half * 8);
  return cat8(lo, hi);
}
// B fragment (32x16 bf16): K-contiguous storage [N][K]; caller passes &Bt[col][kb].
// lane = column, elements 0..7 <- K = kb + half*8 + e, 8..15 <- K = kb + 16 + half*8 + (e-8)
__device__ __forceinline__ bf16x16 fragB(const bf16* colk, int half) {
  bf16x8 lo = *(const bf16x8*)(colk + half * 8);
  bf16x8 hi = *(const bf16x8*)(colk + 16 + half * 8);
  return cat8(lo, hi);
}
__device__ __forceinline__ f32x8 wmma_bf16(bf16x16 a, bf16x16 b, f32x8 c) {
  return __builtin_amdgcn_wmma_f32_16x16x32_bf16(false, a, false, b, (short)0, c,
                                                 false, false);
}

// ---------------- weight fp32 -> bf16 prep ----------------
__global__ void cvt_bf16(const float* __restrict__ in, bf16* __restrict__ o, int n) {
  int i = blockIdx.x * blockDim.x + threadIdx.x;
  if (i < n) o[i] = (bf16)in[i];
}

// ---------------- fused qkv + local attention + proj ----------------
// one workgroup = one (batch, 64-token chunk); 9 waves
__global__ void __launch_bounds__(NTHREADS)
fused_local_attn(const float* __restrict__ x, const bf16* __restrict__ wqkv,
                 const bf16* __restrict__ wproj, const float* __restrict__ pbias,
                 float* __restrict__ out) {
  extern __shared__ __align__(16) char smemraw[];
  float* Ss = (float*)smemraw;            // [64][64]  scores (fp32)
  bf16*  Xs = (bf16*)(Ss + 64 * 64);      // [64][LDX] x tile, token-major
  bf16*  Qt = Xs + 64 * LDX;              // [64][64]  Q token-major (hd padded->64)
  bf16*  Kt = Qt + 64 * 64;               // [64][64]  K token-major
  bf16*  Vd = Kt + 64 * 64;               // [64][64]  V d-major (K=token contiguous)
  bf16*  Ps = Vd + 64 * 64;               // [64][64]  softmax probs
  bf16*  At = Ps + 64 * 64;               // [64][LDX] attention out, token-major

  const int tid  = threadIdx.x;
  const int w    = tid >> 5;
  const int lane = tid & 31;
  const int l15  = lane & 15;
  const int half = lane >> 4;

  const int bb    = blockIdx.x / NC_;
  const int chunk = blockIdx.x % NC_;
  const int n0    = chunk * CH_;
  const float scale = 0.14433756729740643f;   // 48^-0.5, folded into Q

  // ---- load X tile (coalesced over tokens), convert to bf16, token-major ----
  const float* xb = x + (size_t)bb * C_ * N_;
  for (int i = tid; i < C_ * CH_; i += NTHREADS) {
    int n = i & 63, c = i >> 6;
    Xs[n * LDX + c] = (bf16)xb[(size_t)c * N_ + n0 + n];
  }
  // zero Qt/Kt/Vd once (covers the hd 48..63 zero padding; Qt,Kt,Vd contiguous)
  for (int i = tid; i < 3 * 64 * 64; i += NTHREADS) Qt[i] = (bf16)0.0f;
  __syncthreads();

  const int g = w / 3, sub = w % 3;   // g: 0=Q rows,1=K rows,2=V rows; sub: 16-row tile

  for (int h = 0; h < NH_; ++h) {
    // ---- QKV GEMM: this wave computes 16 W-rows x 64 tokens ----
    {
      int wr = g * C_ + h * HD_ + sub * 16;
      f32x8 acc[4] = {};
      for (int kb = 0; kb < C_; kb += 32) {
        bf16x16 a = fragA(wqkv + (size_t)(wr + l15) * C_ + kb, half);
#pragma unroll
        for (int nt = 0; nt < 4; ++nt) {
          bf16x16 b = fragB(Xs + (nt * 16 + l15) * LDX + kb, half);
          acc[nt] = wmma_bf16(a, b, acc[nt]);
        }
      }
      int d0 = sub * 16 + half * 8;  // channel within head for this lane's rows
      if (g == 0) {                  // Q: token-major, pre-scaled
#pragma unroll
        for (int nt = 0; nt < 4; ++nt) {
          int tkn = nt * 16 + l15;
          bf16x8 o;
#pragma unroll
          for (int r = 0; r < 8; ++r) o[r] = (bf16)(acc[nt][r] * scale);
          *(bf16x8*)(Qt + tkn * 64 + d0) = o;
        }
      } else if (g == 1) {           // K: token-major
#pragma unroll
        for (int nt = 0; nt < 4; ++nt) {
          int tkn = nt * 16 + l15;
          bf16x8 o;
#pragma unroll
          for (int r = 0; r < 8; ++r) o[r] = (bf16)acc[nt][r];
          *(bf16x8*)(Kt + tkn * 64 + d0) = o;
        }
      } else {                       // V: d-major (token contiguous per d row)
#pragma unroll
        for (int nt = 0; nt < 4; ++nt) {
          int tkn = nt * 16 + l15;
#pragma unroll
          for (int r = 0; r < 8; ++r) Vd[(d0 + r) * 64 + tkn] = (bf16)acc[nt][r];
        }
      }
    }
    __syncthreads();

    // ---- S = Q · K^T (K-dim = 64, zero-padded hd) : 16 tiles over 9 waves ----
    for (int t = w; t < 16; t += 9) {
      int mi = t >> 2, ni = t & 3;
      f32x8 acc = {};
#pragma unroll
      for (int kb = 0; kb < 64; kb += 32) {
        bf16x16 a = fragA(Qt + (mi * 16 + l15) * 64 + kb, half);
        bf16x16 b = fragB(Kt + (ni * 16 + l15) * 64 + kb, half);
        acc = wmma_bf16(a, b, acc);
      }
#pragma unroll
      for (int r = 0; r < 8; ++r)
        Ss[(mi * 16 + half * 8 + r) * 64 + ni * 16 + l15] = acc[r];
    }
    __syncthreads();

    // ---- row softmax -> Ps (bf16) ----
    if (tid < 64) {
      const float* row = Ss + tid * 64;
      float m = row[0];
      for (int j = 1; j < 64; ++j) m = fmaxf(m, row[j]);
      float s = 0.f;
      for (int j = 0; j < 64; ++j) s += __expf(row[j] - m);
      float inv = 1.f / s;
      for (int j = 0; j < 64; ++j) Ps[tid * 64 + j] = (bf16)(__expf(row[j] - m) * inv);
    }
    __syncthreads();

    // ---- O = P · V : 12 tiles (4 token-tiles x 3 d-tiles) over 9 waves ----
    for (int t = w; t < 12; t += 9) {
      int mi = t / 3, di = t % 3;
      f32x8 acc = {};
#pragma unroll
      for (int kb = 0; kb < 64; kb += 32) {
        bf16x16 a = fragA(Ps + (mi * 16 + l15) * 64 + kb, half);
        bf16x16 b = fragB(Vd + (di * 16 + l15) * 64 + kb, half);
        acc = wmma_bf16(a, b, acc);
      }
      int c = h * HD_ + di * 16 + l15;   // output channel for this lane
#pragma unroll
      for (int r = 0; r < 8; ++r)
        At[(mi * 16 + half * 8 + r) * LDX + c] = (bf16)acc[r];
    }
    __syncthreads();
  }

  // ---- projection: out[b][o][n] = proj_w · At^T + bias ----
  for (int mt = w; mt < 24; mt += 9) {
    f32x8 acc[4] = {};
    for (int kb = 0; kb < C_; kb += 32) {
      bf16x16 a = fragA(wproj + (size_t)(mt * 16 + l15) * C_ + kb, half);
#pragma unroll
      for (int nt = 0; nt < 4; ++nt) {
        bf16x16 b = fragB(At + (nt * 16 + l15) * LDX + kb, half);
        acc[nt] = wmma_bf16(a, b, acc[nt]);
      }
    }
    int o0 = mt * 16 + half * 8;
#pragma unroll
    for (int nt = 0; nt < 4; ++nt) {
      int tkn = n0 + nt * 16 + l15;
#pragma unroll
      for (int r = 0; r < 8; ++r)
        out[((size_t)bb * C_ + o0 + r) * N_ + tkn] = acc[nt][r] + pbias[o0 + r];
    }
  }
}

// ---------------- launch ----------------
extern "C" void kernel_launch(void* const* d_in, const int* in_sizes, int n_in,
                              void* d_out, int out_size, void* d_ws, size_t ws_size,
                              hipStream_t stream) {
  const float* x      = (const float*)d_in[0];
  const float* qkv_w  = (const float*)d_in[1];   // [3C, C]
  const float* proj_w = (const float*)d_in[2];   // [C, C]
  const float* proj_b = (const float*)d_in[3];   // [C]
  float* out = (float*)d_out;                    // [B, C, H, W] = [B, C, N]

  bf16* wq = (bf16*)d_ws;                        // 3C*C bf16
  bf16* wp = wq + 3 * C_ * C_;                   // C*C bf16

  const int nq = 3 * C_ * C_;
  const int np = C_ * C_;
  cvt_bf16<<<(nq + 255) / 256, 256, 0, stream>>>(qkv_w, wq, nq);
  cvt_bf16<<<(np + 255) / 256, 256, 0, stream>>>(proj_w, wp, np);

  const size_t smem =
      64 * 64 * sizeof(float) +                         // Ss
      (size_t)(64 * LDX) * sizeof(bf16) +               // Xs
      (size_t)(3 * 64 * 64) * sizeof(bf16) +            // Qt,Kt,Vd
      (size_t)(64 * 64) * sizeof(bf16) +                // Ps
      (size_t)(64 * LDX) * sizeof(bf16);                // At

  dim3 grid(16 * NC_);   // B * chunks = 1024 workgroups
  fused_local_attn<<<grid, NTHREADS, smem, stream>>>(x, wq, wp, proj_b, out);
}
